// MLP_12317966205169
// MI455X (gfx1250) — matrix operand order repaired and checked by reading, not compiled, not run on hardware
//
#include <hip/hip_runtime.h>

// SwiGLU MLP for MI455X (gfx1250, wave32, WMMA).
//   h   = silu(x @ Wg^T) * (x @ Wu^T)      [8192, 14336] fp16 (in d_ws)
//   out = h @ Wd^T                          [8192, 4096]  fp16
// Memory-bound on weight streaming (352 MB @ 23.3 TB/s ~= 18 us floor).
// fp16 WMMA (v_wmma_f32_16x16x32_f16), f32 accumulation, fused SiLU.
// Per-wave tile: 32 tokens x 32 outputs -> 8 WMMAs share 4 B-tiles per
// K-step (2 WMMAs per weight load) and each block covers 128 tokens,
// minimizing how often each weight element is streamed from HBM/L2.

typedef __attribute__((ext_vector_type(16))) _Float16 v16h;
typedef __attribute__((ext_vector_type(8)))  _Float16 v8h;
typedef __attribute__((ext_vector_type(8)))  float    v8f;

#define HD  4096
#define FF  14336
#define TOK 8192

// ---- A-matrix tile: 16 (rows) x 32 (K), fp16, wave32 ----
// Contiguous-K A 16x32 layout (same k order as load_b_tile): lane L holds row L%16; K chunks at
// (L<16?0:16) and +8 -> two contiguous 16B loads per lane.
__device__ __forceinline__ v16h load_a_tile(const _Float16* __restrict__ base,
                                            int ld, int row0, int k0,
                                            int laneM, int laneHi) {
  const _Float16* p = base + (size_t)(row0 + laneM) * ld + k0 + laneHi * 16;
  v8h lo = *(const v8h*)(p);        // K = k0 + 16*hi + {0..7}
  v8h hi = *(const v8h*)(p + 8);    // K = k0 + 16*hi + 8 + {0..7}
  v16h a;
#pragma unroll
  for (int i = 0; i < 8; ++i) { a[i] = lo[i]; a[i + 8] = hi[i]; }
  return a;
}

// ---- B-matrix tile: 32 (K) x 16 (cols), fp16, wave32 ----
// Lane L holds output column n0 + L%16, 16 contiguous K halves at
// (L<16?0:16) -> one 32B load per lane. Weights are [out,in] row-major.
__device__ __forceinline__ v16h load_b_tile(const _Float16* __restrict__ base,
                                            int ld, int n0, int k0,
                                            int laneN, int laneHi) {
  const _Float16* p = base + (size_t)(n0 + laneN) * ld + k0 + laneHi * 16;
  return *(const v16h*)(p);
}

__device__ __forceinline__ float silu_f32(float g) {
  return g / (1.0f + __expf(-g));
}

#define WMMA_F16(A, B, C) \
  __builtin_amdgcn_wmma_f32_16x16x32_f16(false, (A), false, (B), (short)0, (C), false, false)

// ============================================================
// Kernel 1: fused gate/up GEMM + SiLU.
// Block = 256 thr = 8 waves, arranged 4 (token) x 2 (ffn).
// Wave tile = 32 tok x 32 ffn; block tile = 128 tok x 64 ffn.
// Grid = (FF/64, TOK/128) = (224, 64).
// ============================================================
__global__ __launch_bounds__(256)
void swiglu_gateup_kernel(const _Float16* __restrict__ x,
                          const _Float16* __restrict__ Wg,
                          const _Float16* __restrict__ Wu,
                          _Float16* __restrict__ h) {
  const int lane   = threadIdx.x & 31;
  const int wave   = threadIdx.x >> 5;   // 0..7
  const int waveT  = wave & 3;           // token-dim wave
  const int waveN  = wave >> 2;          // ffn-dim wave
  const int laneM  = lane & 15;
  const int laneHi = lane >> 4;

  const int t0 = blockIdx.y * 128 + waveT * 32;
  const int n0 = blockIdx.x * 64 + waveN * 32;

  v8f ag00 = {}, ag01 = {}, ag10 = {}, ag11 = {};
  v8f au00 = {}, au01 = {}, au10 = {}, au11 = {};

#pragma unroll 2
  for (int k0 = 0; k0 < HD; k0 += 32) {
    // Unconditional speculative prefetch of next K-chunk of streaming
    // weight rows (global_prefetch_b8; OOB translations silently dropped).
    __builtin_prefetch(Wg + (size_t)(n0 + laneM) * HD + k0 + 64, 0, 1);
    __builtin_prefetch(Wu + (size_t)(n0 + laneM) * HD + k0 + 64, 0, 1);

    v16h a0  = load_a_tile(x, HD, t0,      k0, laneM, laneHi);
    v16h a1  = load_a_tile(x, HD, t0 + 16, k0, laneM, laneHi);
    v16h bg0 = load_b_tile(Wg, HD, n0,      k0, laneM, laneHi);
    v16h bg1 = load_b_tile(Wg, HD, n0 + 16, k0, laneM, laneHi);
    v16h bu0 = load_b_tile(Wu, HD, n0,      k0, laneM, laneHi);
    v16h bu1 = load_b_tile(Wu, HD, n0 + 16, k0, laneM, laneHi);

    // 8 WMMAs: each B tile feeds 2, each A tile feeds 4.
    ag00 = WMMA_F16(a0, bg0, ag00);
    ag01 = WMMA_F16(a0, bg1, ag01);
    ag10 = WMMA_F16(a1, bg0, ag10);
    ag11 = WMMA_F16(a1, bg1, ag11);
    au00 = WMMA_F16(a0, bu0, au00);
    au01 = WMMA_F16(a0, bu1, au01);
    au10 = WMMA_F16(a1, bu0, au10);
    au11 = WMMA_F16(a1, bu1, au11);
  }

  // Fused SiLU(gate)*up epilogue. D layout: VGPR r, lane ->
  // row r + 8*laneHi, col laneM.
#pragma unroll
  for (int r = 0; r < 8; ++r) {
    const int ta = t0 + r + laneHi * 8;
    const int tb = ta + 16;
    float h00 = silu_f32(ag00[r]) * au00[r];
    float h01 = silu_f32(ag01[r]) * au01[r];
    float h10 = silu_f32(ag10[r]) * au10[r];
    float h11 = silu_f32(ag11[r]) * au11[r];
    h[(size_t)ta * FF + n0 + laneM]      = (_Float16)h00;
    h[(size_t)ta * FF + n0 + 16 + laneM] = (_Float16)h01;
    h[(size_t)tb * FF + n0 + laneM]      = (_Float16)h10;
    h[(size_t)tb * FF + n0 + 16 + laneM] = (_Float16)h11;
  }
}

// ============================================================
// Kernel 2: down projection  out = h @ Wd^T, K = 14336.
// Wave tile 32 tok x 32 hd; block tile 128 x 64.
// Grid = (HD/64, TOK/128) = (64, 64).
// ============================================================
__global__ __launch_bounds__(256)
void down_proj_kernel(const _Float16* __restrict__ h,
                      const _Float16* __restrict__ Wd,
                      _Float16* __restrict__ out) {
  const int lane   = threadIdx.x & 31;
  const int wave   = threadIdx.x >> 5;
  const int waveT  = wave & 3;
  const int waveN  = wave >> 2;
  const int laneM  = lane & 15;
  const int laneHi = lane >> 4;

  const int t0 = blockIdx.y * 128 + waveT * 32;
  const int n0 = blockIdx.x * 64 + waveN * 32;

  v8f a00 = {}, a01 = {}, a10 = {}, a11 = {};

#pragma unroll 2
  for (int k0 = 0; k0 < FF; k0 += 32) {
    __builtin_prefetch(Wd + (size_t)(n0 + laneM) * FF + k0 + 64, 0, 1);

    v16h t_a0 = load_a_tile(h, FF, t0,      k0, laneM, laneHi);
    v16h t_a1 = load_a_tile(h, FF, t0 + 16, k0, laneM, laneHi);
    v16h b0   = load_b_tile(Wd, FF, n0,      k0, laneM, laneHi);
    v16h b1   = load_b_tile(Wd, FF, n0 + 16, k0, laneM, laneHi);

    a00 = WMMA_F16(t_a0, b0, a00);
    a01 = WMMA_F16(t_a0, b1, a01);
    a10 = WMMA_F16(t_a1, b0, a10);
    a11 = WMMA_F16(t_a1, b1, a11);
  }

#pragma unroll
  for (int r = 0; r < 8; ++r) {
    const int ta = t0 + r + laneHi * 8;
    const int tb = ta + 16;
    out[(size_t)ta * HD + n0 + laneM]      = (_Float16)a00[r];
    out[(size_t)ta * HD + n0 + 16 + laneM] = (_Float16)a01[r];
    out[(size_t)tb * HD + n0 + laneM]      = (_Float16)a10[r];
    out[(size_t)tb * HD + n0 + 16 + laneM] = (_Float16)a11[r];
  }
}

extern "C" void kernel_launch(void* const* d_in, const int* in_sizes, int n_in,
                              void* d_out, int out_size, void* d_ws, size_t ws_size,
                              hipStream_t stream) {
  const _Float16* x  = (const _Float16*)d_in[0];  // [8192, 4096]
  const _Float16* Wg = (const _Float16*)d_in[1];  // [14336, 4096]
  const _Float16* Wu = (const _Float16*)d_in[2];  // [14336, 4096]
  const _Float16* Wd = (const _Float16*)d_in[3];  // [4096, 14336]
  _Float16* out  = (_Float16*)d_out;              // [8192, 4096]
  _Float16* hbuf = (_Float16*)d_ws;               // [8192, 14336] fp16 = 235 MB

  dim3 block(256);
  dim3 grid1(FF / 64, TOK / 128);   // 224 x 64
  swiglu_gateup_kernel<<<grid1, block, 0, stream>>>(x, Wg, Wu, hbuf);

  dim3 grid2(HD / 64, TOK / 128);   // 64 x 64
  down_proj_kernel<<<grid2, block, 0, stream>>>(hbuf, Wd, out);
}
